// FlashAttention_26860725469380
// MI455X (gfx1250) — compile-verified, matrix operand order checked by reading, not run on hardware
//
#include <hip/hip_runtime.h>

// ---------------------------------------------------------------------------
// CDNA5 (gfx1250) fused causal multi-head attention, bf16 WMMA everywhere.
// D = A*B + C via v_wmma_f32_16x16x32_bf16 (wave32, 16x16 tiles, K=32).
// ---------------------------------------------------------------------------

typedef __bf16 bf16;
typedef __attribute__((ext_vector_type(8)))  __bf16 v8bf;
typedef __attribute__((ext_vector_type(16))) __bf16 v16bf;
typedef __attribute__((ext_vector_type(8)))  float  v8f;

#define WMMA_BF16(A, B, C) \
  __builtin_amdgcn_wmma_f32_16x16x32_bf16(false, (A), false, (B), (short)0, (C), false, false)

// Per ISA 7.12.2 (16-bit A 16x32): lane half selects K-base 0/8; each lane
// holds K = base..base+7 (8 bf16 = 16B) and base+16..base+23 (16B).
__device__ __forceinline__ v16bf make_frag(const bf16* lo, const bf16* hi) {
  v8bf l = *(const v8bf*)lo;
  v8bf h = *(const v8bf*)hi;
  v16bf r;
#pragma unroll
  for (int i = 0; i < 8; ++i) { r[i] = l[i]; r[i + 8] = h[i]; }
  return r;
}

__device__ __forceinline__ v8f vzero8() {
  v8f z;
#pragma unroll
  for (int i = 0; i < 8; ++i) z[i] = 0.0f;
  return z;
}

// ---------------------------------------------------------------------------
// Kernel 1: QKV projection. X[8192,1024] (f32) @ W[1024,3072] (f32) -> bf16
// q,k,v in [B,H,S,D] layout. Scale folded into q. 128x128 block tile,
// 8 waves of 32x64 (2x4 WMMA accumulators each), K step 32.
// ---------------------------------------------------------------------------
__global__ __launch_bounds__(256) void qkv_gemm_kernel(
    const float* __restrict__ X, const float* __restrict__ W,
    bf16* __restrict__ Qo, bf16* __restrict__ Ko, bf16* __restrict__ Vo) {
  __shared__ bf16 As[128 * 40];  // [row][k], pitch 40 (16B-aligned rows)
  __shared__ bf16 Bs[128 * 40];  // [n][k] transposed, pitch 40
  const int tid = threadIdx.x;
  const int lane = tid & 31;
  const int wv = tid >> 5;
  const int wm = wv >> 1;        // 0..3: wave row
  const int wn = wv & 1;         // 0..1: wave col
  const int m0 = blockIdx.y * 128;
  const int n0 = blockIdx.x * 128;
  const int l15 = lane & 15;
  const int klo = (lane < 16) ? 0 : 8;
  const int hb = (lane >> 4) << 3;  // 0 or 8: row offset of this lane half in C/D

  v8f acc[2][4];
#pragma unroll
  for (int i = 0; i < 2; ++i)
#pragma unroll
    for (int j = 0; j < 4; ++j) acc[i][j] = vzero8();

  for (int k0 = 0; k0 < 1024; k0 += 32) {
    __syncthreads();
    // Stage A tile (convert f32 -> bf16)
#pragma unroll
    for (int t = 0; t < 4; ++t) {
      int idx = tid + t * 256;
      int row = idx >> 3;
      int cv = (idx & 7) << 2;
      float4 f = *(const float4*)(X + (size_t)(m0 + row) * 1024 + k0 + cv);
      bf16* dst = &As[row * 40 + cv];
      dst[0] = (bf16)f.x; dst[1] = (bf16)f.y; dst[2] = (bf16)f.z; dst[3] = (bf16)f.w;
    }
    // Stage B tile transposed (so per-column K is contiguous)
#pragma unroll
    for (int t = 0; t < 4; ++t) {
      int idx = tid + t * 256;
      int kr = idx >> 5;
      int nv = (idx & 31) << 2;
      float4 f = *(const float4*)(W + (size_t)(k0 + kr) * 3072 + n0 + nv);
      Bs[(nv + 0) * 40 + kr] = (bf16)f.x;
      Bs[(nv + 1) * 40 + kr] = (bf16)f.y;
      Bs[(nv + 2) * 40 + kr] = (bf16)f.z;
      Bs[(nv + 3) * 40 + kr] = (bf16)f.w;
    }
    __syncthreads();
    v16bf a[2], b[4];
#pragma unroll
    for (int mi = 0; mi < 2; ++mi) {
      int row = wm * 32 + mi * 16 + l15;
      a[mi] = make_frag(&As[row * 40 + klo], &As[row * 40 + klo + 16]);
    }
#pragma unroll
    for (int ni = 0; ni < 4; ++ni) {
      int col = wn * 64 + ni * 16 + l15;
      b[ni] = make_frag(&Bs[col * 40 + klo], &Bs[col * 40 + klo + 16]);
    }
#pragma unroll
    for (int mi = 0; mi < 2; ++mi)
#pragma unroll
      for (int ni = 0; ni < 4; ++ni)
        acc[mi][ni] = WMMA_BF16(a[mi], b[ni], acc[mi][ni]);
  }

  // Epilogue: scatter into q/k/v [B,H,S,D] as bf16, fold softmax scale into q.
#pragma unroll
  for (int mi = 0; mi < 2; ++mi)
#pragma unroll
    for (int ni = 0; ni < 4; ++ni) {
      int n = n0 + wn * 64 + ni * 16 + l15;
      int which = n >> 10;     // 0=q 1=k 2=v
      int c = n & 1023;
      int h = c >> 6;
      int d = c & 63;
#pragma unroll
      for (int r = 0; r < 8; ++r) {
        int m = m0 + wm * 32 + mi * 16 + r + hb;
        int bidx = m >> 11;    // batch
        int s = m & 2047;      // seq
        size_t off = ((size_t)(bidx * 16 + h) * 2048 + s) * 64 + d;
        float val = acc[mi][ni][r];
        if (which == 0)      Qo[off] = (bf16)(val * 0.125f);  // 64^-0.5
        else if (which == 1) Ko[off] = (bf16)val;
        else                 Vo[off] = (bf16)val;
      }
    }
}

// ---------------------------------------------------------------------------
// Kernel 2: causal flash attention. Block = (b*h, 128-query tile), 8 waves,
// each wave owns 16 query rows; Q fragments resident in registers.
// KV streamed in 32-key LDS blocks; online softmax; P round-trips through
// per-wave LDS to reshape accumulator -> A fragment.
// ---------------------------------------------------------------------------
__global__ __launch_bounds__(256) void flash_attn_kernel(
    const bf16* __restrict__ Q, const bf16* __restrict__ K,
    const bf16* __restrict__ V, bf16* __restrict__ O) {
  __shared__ bf16 Ks[32 * 72];       // [key][d], pitch 72
  __shared__ bf16 Vt[64 * 40];       // [d][key] transposed, pitch 40
  __shared__ bf16 Ps[8 * 16 * 40];   // per-wave P tile [16][32], pitch 40
  const int tid = threadIdx.x;
  const int lane = tid & 31;
  const int wv = tid >> 5;
  const int bh = blockIdx.y;         // 0..63
  const int qtile = blockIdx.x;      // 0..15
  const int qbase = qtile * 128 + wv * 16;
  const int l15 = lane & 15;
  const int klo = (lane < 16) ? 0 : 8;
  const int hb = (lane >> 4) << 3;

  const bf16* Qp = Q + (size_t)bh * 2048 * 64;
  const bf16* Kp = K + (size_t)bh * 2048 * 64;
  const bf16* Vp = V + (size_t)bh * 2048 * 64;

  // Q A-fragments for d=0..31 and d=32..63 (already pre-scaled by 1/8)
  v16bf aQ[2];
  {
    const bf16* qr = Qp + (size_t)(qbase + l15) * 64;
    aQ[0] = make_frag(qr + klo, qr + klo + 16);
    aQ[1] = make_frag(qr + 32 + klo, qr + 32 + klo + 16);
  }

  v8f accO[4];
#pragma unroll
  for (int ni = 0; ni < 4; ++ni) accO[ni] = vzero8();
  float mrow[8], lrow[8];
#pragma unroll
  for (int r = 0; r < 8; ++r) { mrow[r] = -1e30f; lrow[r] = 0.0f; }

  const int nkb = qtile * 4 + 4;  // causal: keys up to end of this q tile
  bf16* pw = &Ps[wv * 16 * 40];

  for (int kb = 0; kb < nkb; ++kb) {
    __syncthreads();
    // Cooperative stage: K row-major, V transposed
    {
      int key = tid >> 3;
      int d = (tid & 7) << 3;
      size_t goff = (size_t)(kb * 32 + key) * 64 + d;
      v8bf kd = *(const v8bf*)(Kp + goff);
      *(v8bf*)&Ks[key * 72 + d] = kd;
      v8bf vd = *(const v8bf*)(Vp + goff);
#pragma unroll
      for (int j = 0; j < 8; ++j) Vt[(d + j) * 40 + key] = vd[j];
    }
    __syncthreads();

    if (kb * 32 <= qbase + 15) {  // wave-uniform: any unmasked key in block?
      // S = Q @ K^T  (two 16-key tiles, K-dim = head dim = 64)
      v8f accS[2];
      accS[0] = vzero8(); accS[1] = vzero8();
#pragma unroll
      for (int ni = 0; ni < 2; ++ni) {
        const bf16* kr = &Ks[(ni * 16 + l15) * 72];
#pragma unroll
        for (int db = 0; db < 2; ++db) {
          v16bf bK = make_frag(kr + db * 32 + klo, kr + db * 32 + klo + 16);
          accS[ni] = WMMA_BF16(aQ[db], bK, accS[ni]);
        }
      }
      // Causal mask + online softmax (rows live on 16-lane halves)
      float alpha[8];
#pragma unroll
      for (int r = 0; r < 8; ++r) {
        const int qrow = qbase + r + hb;
#pragma unroll
        for (int ni = 0; ni < 2; ++ni) {
          int kcol = kb * 32 + ni * 16 + l15;
          if (kcol > qrow) accS[ni][r] = -1e30f;
        }
        float mx = fmaxf(accS[0][r], accS[1][r]);
#pragma unroll
        for (int off = 1; off < 16; off <<= 1)
          mx = fmaxf(mx, __shfl_xor(mx, off));
        float nm = fmaxf(mrow[r], mx);
        alpha[r] = __expf(mrow[r] - nm);
        mrow[r] = nm;
        float p0 = __expf(accS[0][r] - nm);
        float p1 = __expf(accS[1][r] - nm);
        accS[0][r] = p0; accS[1][r] = p1;
        float rs = p0 + p1;
#pragma unroll
        for (int off = 1; off < 16; off <<= 1)
          rs += __shfl_xor(rs, off);
        lrow[r] = lrow[r] * alpha[r] + rs;
      }
#pragma unroll
      for (int ni = 0; ni < 4; ++ni)
#pragma unroll
        for (int r = 0; r < 8; ++r) accO[ni][r] *= alpha[r];

      // Reshape P: C/D layout -> A layout via per-wave LDS (LDS is in-order
      // within a wave; add explicit dscnt wait before the dependent reads).
#pragma unroll
      for (int r = 0; r < 8; ++r) {
        pw[(r + hb) * 40 + l15]      = (bf16)accS[0][r];
        pw[(r + hb) * 40 + 16 + l15] = (bf16)accS[1][r];
      }
      asm volatile("s_wait_dscnt 0" ::: "memory");
      v16bf aP = make_frag(&pw[l15 * 40 + klo], &pw[l15 * 40 + klo + 16]);
      // O += P @ V  (K-dim = 32 keys; B columns = head dims, contiguous in Vt)
#pragma unroll
      for (int ni = 0; ni < 4; ++ni) {
        const bf16* vr = &Vt[(ni * 16 + l15) * 40];
        v16bf bV = make_frag(vr + klo, vr + klo + 16);
        accO[ni] = WMMA_BF16(aP, bV, accO[ni]);
      }
    }
  }

  // Epilogue: normalize and store to attn-out [B,S,C] bf16
  const int b = bh >> 4;
  const int h = bh & 15;
#pragma unroll
  for (int r = 0; r < 8; ++r) {
    float inv = 1.0f / lrow[r];
    int s = qbase + r + hb;
    bf16* orow = O + ((size_t)(b * 2048 + s)) * 1024 + h * 64;
#pragma unroll
    for (int ni = 0; ni < 4; ++ni)
      orow[ni * 16 + l15] = (bf16)(accO[ni][r] * inv);
  }
}

// ---------------------------------------------------------------------------
// Kernel 3: output projection. A[8192,1024] bf16 @ W[1024,1024] f32 + bias,
// fp32 output. Same tiling as kernel 1.
// ---------------------------------------------------------------------------
__global__ __launch_bounds__(256) void proj_gemm_kernel(
    const bf16* __restrict__ A, const float* __restrict__ W,
    const float* __restrict__ bias, float* __restrict__ out) {
  __shared__ bf16 As[128 * 40];
  __shared__ bf16 Bs[128 * 40];
  const int tid = threadIdx.x;
  const int lane = tid & 31;
  const int wv = tid >> 5;
  const int wm = wv >> 1;
  const int wn = wv & 1;
  const int m0 = blockIdx.y * 128;
  const int n0 = blockIdx.x * 128;
  const int l15 = lane & 15;
  const int klo = (lane < 16) ? 0 : 8;
  const int hb = (lane >> 4) << 3;

  v8f acc[2][4];
#pragma unroll
  for (int i = 0; i < 2; ++i)
#pragma unroll
    for (int j = 0; j < 4; ++j) acc[i][j] = vzero8();

  for (int k0 = 0; k0 < 1024; k0 += 32) {
    __syncthreads();
#pragma unroll
    for (int t = 0; t < 2; ++t) {  // A already bf16: 2x b128 per thread
      int idx = tid + t * 256;
      int row = idx >> 2;
      int ch = (idx & 3) << 3;
      v8bf va = *(const v8bf*)(A + (size_t)(m0 + row) * 1024 + k0 + ch);
      *(v8bf*)&As[row * 40 + ch] = va;
    }
#pragma unroll
    for (int t = 0; t < 4; ++t) {
      int idx = tid + t * 256;
      int kr = idx >> 5;
      int nv = (idx & 31) << 2;
      float4 f = *(const float4*)(W + (size_t)(k0 + kr) * 1024 + n0 + nv);
      Bs[(nv + 0) * 40 + kr] = (bf16)f.x;
      Bs[(nv + 1) * 40 + kr] = (bf16)f.y;
      Bs[(nv + 2) * 40 + kr] = (bf16)f.z;
      Bs[(nv + 3) * 40 + kr] = (bf16)f.w;
    }
    __syncthreads();
    v16bf a[2], b[4];
#pragma unroll
    for (int mi = 0; mi < 2; ++mi) {
      int row = wm * 32 + mi * 16 + l15;
      a[mi] = make_frag(&As[row * 40 + klo], &As[row * 40 + klo + 16]);
    }
#pragma unroll
    for (int ni = 0; ni < 4; ++ni) {
      int col = wn * 64 + ni * 16 + l15;
      b[ni] = make_frag(&Bs[col * 40 + klo], &Bs[col * 40 + klo + 16]);
    }
#pragma unroll
    for (int mi = 0; mi < 2; ++mi)
#pragma unroll
      for (int ni = 0; ni < 4; ++ni)
        acc[mi][ni] = WMMA_BF16(a[mi], b[ni], acc[mi][ni]);
  }

#pragma unroll
  for (int ni = 0; ni < 4; ++ni) {
    int n = n0 + wn * 64 + ni * 16 + l15;
    float bn = bias[n];
#pragma unroll
    for (int mi = 0; mi < 2; ++mi)
#pragma unroll
      for (int r = 0; r < 8; ++r) {
        int m = m0 + wm * 32 + mi * 16 + r + hb;
        out[(size_t)m * 1024 + n] = acc[mi][ni][r] + bn;
      }
  }
}

// ---------------------------------------------------------------------------
extern "C" void kernel_launch(void* const* d_in, const int* in_sizes, int n_in,
                              void* d_out, int out_size, void* d_ws, size_t ws_size,
                              hipStream_t stream) {
  (void)in_sizes; (void)n_in; (void)out_size; (void)ws_size;
  const float* x      = (const float*)d_in[0];
  const float* w_qkv  = (const float*)d_in[1];
  const float* w_proj = (const float*)d_in[2];
  const float* b_proj = (const float*)d_in[3];
  float* out = (float*)d_out;

  const size_t elems = (size_t)8192 * 1024;  // B*S * DIM
  bf16* q  = (bf16*)d_ws;        // 16 MB each
  bf16* k  = q + elems;
  bf16* v  = k + elems;
  bf16* ao = v + elems;          // total 64 MB workspace

  // QKV: M=8192, N=3072 -> grid (24, 64)
  qkv_gemm_kernel<<<dim3(24, 64), 256, 0, stream>>>(x, w_qkv, q, k, v);
  // Attention: 16 query tiles x 64 (b,h) pairs
  flash_attn_kernel<<<dim3(16, 64), 256, 0, stream>>>(q, k, v, ao);
  // Projection: M=8192, N=1024 -> grid (8, 64)
  proj_gemm_kernel<<<dim3(8, 64), 256, 0, stream>>>(ao, w_proj, b_proj, out);
}
